// LNN_23356032155903
// MI455X (gfx1250) — matrix operand order repaired and checked by reading, not compiled, not run on hardware
//
#include <hip/hip_runtime.h>

// ---------------------------------------------------------------------------
// CfC liquid-NN scan for MI455X (gfx1250, wave32, WMMA bf16 16x16x32).
// B=256, S=512, I=256, H=512, BB=256, O=128, depth=4.
// 16 workgroups (one 16-row batch tile each) x 16 waves (512 threads).
// Weights pre-packed to bf16 B-fragment layout in d_ws by pack_bfrag().
// opaque_i() pins weight streaming inside the t-loop (defeats LICM spill)
// while preserving addrspace(1) so loads lower to global_load_b128, not flat.
// ---------------------------------------------------------------------------

typedef __attribute__((ext_vector_type(16))) __bf16 v16bf;
typedef __attribute__((ext_vector_type(8)))  __bf16 v8bf;
typedef __attribute__((ext_vector_type(8)))  float  v8f;

#define SEQ   512
#define INP   256
#define HID   512
#define BBD   256
#define OUTD  128

#define AB_STRIDE 776   // 768 + 8 pad (bf16 elems) -> 1552B rows, bank-staggered
#define Z_STRIDE  264   // 256 + 8 pad

__device__ __forceinline__ __bf16 f2bf(float f) {
    unsigned u = __builtin_bit_cast(unsigned, f);
    u += 0x7FFFu + ((u >> 16) & 1u);               // round-to-nearest-even
    unsigned short h = (unsigned short)(u >> 16);
    return __builtin_bit_cast(__bf16, h);
}

__device__ __forceinline__ float tanh_fast(float x) {
    float e = __expf(2.0f * x);                    // v_exp_f32 path
    return 1.0f - 2.0f / (e + 1.0f);
}
__device__ __forceinline__ float sigmoid_fast(float x) {
    return 1.0f / (1.0f + __expf(-x));
}

// Opaque no-op on a 32-bit offset.  Executed inside the scan loop it makes
// the weight addresses non-loop-invariant (asm volatile = black box), so LLVM
// cannot hoist the dependent loads out of the 512-step loop and spill them to
// scratch.  Crucially the *pointer* itself is untouched, so address-space
// inference still proves it global -> global_load_b128 (LOADcnt only), not
// flat_load_b128 (which would also tie up DScnt / the LDS pipe we need for
// A-fragment ds_loads).
__device__ __forceinline__ int opaque_i(int v) {
    asm volatile("" : "+v"(v));
    return v;
}

// A fragment (16x32 bf16, MxK) from LDS: p already points at
// row(l&15)*stride + k0 + ((lane>>4)*8).  Two 16B chunks 32B apart in K.
__device__ __forceinline__ v16bf ldsA(const __bf16* p) {
    v8bf lo = *(const v8bf*)p;
    v8bf hi = *(const v8bf*)(p + 16);
    return __builtin_shufflevector(lo, hi,
        0, 1, 2, 3, 4, 5, 6, 7, 8, 9, 10, 11, 12, 13, 14, 15);
}

// B fragment (32x16 bf16, KxN): pre-packed 32B per lane, contiguous.
__device__ __forceinline__ v16bf ldgB(const __bf16* p) {
    return *(const v16bf*)p;
}

#define WMMA_BF16(a, b, c) \
    __builtin_amdgcn_wmma_f32_16x16x32_bf16(false, (a), false, (b), (short)0, (c), false, false)

// ---------------------------------------------------------------------------
// Pack fp32 row-major W[K][N] into bf16 WMMA-B fragment tiles.
// Tile = 32 lanes x 16 bf16.  lane: n = nt*16 + (lane&15),
// K = kt*32 + (e>>3)*16 + (lane>>4)*8 + (e&7)  (table order, matches ldsA).
// Tiles ordered nt-major, kt-minor (streaming along K per wave).
// ---------------------------------------------------------------------------
__global__ void pack_bfrag(const float* __restrict__ src, __bf16* __restrict__ dst,
                           int N, int numKt, int total) {
    int idx = blockIdx.x * blockDim.x + threadIdx.x;
    if (idx >= total) return;
    int tile = idx >> 9;
    int r    = idx & 511;
    int lane = r >> 4;
    int e    = r & 15;
    int kt   = tile % numKt;
    int nt   = tile / numKt;
    int k    = kt * 32 + ((e >> 3) << 4) + ((lane >> 4) << 3) + (e & 7);
    int n    = (nt << 4) + (lane & 15);
    dst[idx] = f2bf(src[k * N + n]);
}

// ---------------------------------------------------------------------------
// Main scan kernel: grid = 16 (batch tiles), block = 512 (16 wave32s).
// ---------------------------------------------------------------------------
__global__ void __launch_bounds__(512, 1) cfc_scan_kernel(
    const float* __restrict__ x,   const float* __restrict__ ts,
    const __bf16* __restrict__ wb0, const __bf16* __restrict__ wbs,
    const __bf16* __restrict__ wf1, const __bf16* __restrict__ wf2,
    const __bf16* __restrict__ wta, const __bf16* __restrict__ wtb,
    const __bf16* __restrict__ wout,
    const float* __restrict__ bb0, const float* __restrict__ bbs,
    const float* __restrict__ bf1, const float* __restrict__ bf2,
    const float* __restrict__ bta, const float* __restrict__ btb,
    const float* __restrict__ bout,
    float* __restrict__ out)
{
    __shared__ __align__(16) __bf16 Abuf[16 * AB_STRIDE]; // [x(0..255) | h(256..767)]
    __shared__ __align__(16) __bf16 zA[16 * Z_STRIDE];
    __shared__ __align__(16) __bf16 zB[16 * Z_STRIDE];
    __shared__ float bias[3200];   // bb0|bbs(3x256)|bf1|bf2|bta|btb|bout
    __shared__ float tsrow[16];

    const int tid  = threadIdx.x;
    const int w    = tid >> 5;          // wave 0..15
    const int lane = tid & 31;
    const int l15  = lane & 15;
    const int h8   = (lane >> 4) << 3;  // 0 or 8 (K-half for A, M-half for C)
    const int b0   = blockIdx.x << 4;   // first batch row of this WG

    const v8f vzero = {0.f, 0.f, 0.f, 0.f, 0.f, 0.f, 0.f, 0.f};

    // ---- one-time: biases -> LDS, h0 = 0 ----
    for (int j = tid; j < 256; j += 512) bias[j]        = bb0[j];
    for (int j = tid; j < 768; j += 512) bias[256 + j]  = bbs[j];
    for (int j = tid; j < 512; j += 512) bias[1024 + j] = bf1[j];
    for (int j = tid; j < 512; j += 512) bias[1536 + j] = bf2[j];
    for (int j = tid; j < 512; j += 512) bias[2048 + j] = bta[j];
    for (int j = tid; j < 512; j += 512) bias[2560 + j] = btb[j];
    for (int j = tid; j < 128; j += 512) bias[3072 + j] = bout[j];
    {
        const __bf16 zb = f2bf(0.0f);
        for (int q = tid; q < 16 * 512; q += 512) {
            int m = q >> 9, c = q & 511;
            Abuf[m * AB_STRIDE + 256 + c] = zb;
        }
    }

    for (int t = 0; t < SEQ; ++t) {
        // ---- stage x_t (16x256 f32 -> bf16) and ts into LDS ----
        {
            int q = tid << 3;                 // 8 elems/thread
            int m = q >> 8;
            int i = q & 255;
            const float* xp = x + (((size_t)(b0 + m) * SEQ + t) << 8) + i;
            float4 a = *(const float4*)xp;
            float4 b = *(const float4*)(xp + 4);
            v8bf vv;
            vv[0] = f2bf(a.x); vv[1] = f2bf(a.y); vv[2] = f2bf(a.z); vv[3] = f2bf(a.w);
            vv[4] = f2bf(b.x); vv[5] = f2bf(b.y); vv[6] = f2bf(b.z); vv[7] = f2bf(b.w);
            *(v8bf*)&Abuf[m * AB_STRIDE + i] = vv;
            if (tid < 16) tsrow[tid] = ts[(size_t)(b0 + tid) * SEQ + t];
        }
        __syncthreads();

        // ---- L0: [x|h](16x768) @ Wb0(768x256) -> zA, tanh. 1 N-tile/wave ----
        {
            v8f c = vzero;
            const __bf16* arow = Abuf + l15 * AB_STRIDE + h8;
            const __bf16* wp   = wb0 + opaque_i(((w * 24) << 9) + (lane << 4));
            #pragma unroll
            for (int kt = 0; kt < 24; ++kt) {
                v16bf a = ldsA(arow + kt * 32);
                v16bf b = ldgB(wp + (kt << 9));
                c = WMMA_BF16(a, b, c);
            }
            float bs = bias[w * 16 + l15];
            #pragma unroll
            for (int v = 0; v < 8; ++v)
                zA[(v + h8) * Z_STRIDE + w * 16 + l15] = f2bf(tanh_fast(c[v] + bs));
        }
        __syncthreads();

        // ---- L1..L3: z(16x256) @ Wbs[li](256x256), ping-pong zA<->zB ----
        __bf16* zin  = zA;
        __bf16* zout = zB;
        #pragma unroll
        for (int li = 0; li < 3; ++li) {
            v8f c = vzero;
            const __bf16* arow = zin + l15 * Z_STRIDE + h8;
            const __bf16* wp   = wbs + (li << 16) + opaque_i(((w * 8) << 9) + (lane << 4));
            #pragma unroll
            for (int kt = 0; kt < 8; ++kt) {
                v16bf a = ldsA(arow + kt * 32);
                v16bf b = ldgB(wp + (kt << 9));
                c = WMMA_BF16(a, b, c);
            }
            float bs = bias[256 + (li << 8) + w * 16 + l15];
            #pragma unroll
            for (int v = 0; v < 8; ++v)
                zout[(v + h8) * Z_STRIDE + w * 16 + l15] = f2bf(tanh_fast(c[v] + bs));
            __syncthreads();
            __bf16* tmp = zin; zin = zout; zout = tmp;
        }
        // zin now holds z3 (== zB)

        // ---- heads: z @ {Wff1,Wff2,Wta,Wtb}(256x512); wave owns 32 h-cols ----
        {
            v8f c1[2] = {vzero, vzero}, c2[2] = {vzero, vzero};
            v8f ca[2] = {vzero, vzero}, cb[2] = {vzero, vzero};
            const __bf16* arow = zin + l15 * Z_STRIDE + h8;
            // per-step opaque offsets: loads stay in-loop, stream from L2
            int hoff = opaque_i(((w * 16) << 9) + (lane << 4));
            const __bf16* p1 = wf1 + hoff;
            const __bf16* p2 = wf2 + hoff;
            const __bf16* pa = wta + hoff;
            const __bf16* pb = wtb + hoff;
            #pragma unroll
            for (int kt = 0; kt < 8; ++kt) {
                v16bf a = ldsA(arow + kt * 32);   // reused across all 8 WMMAs
                #pragma unroll
                for (int j = 0; j < 2; ++j) {
                    int toff = ((j * 8 + kt) << 9);
                    c1[j] = WMMA_BF16(a, ldgB(p1 + toff), c1[j]);
                    c2[j] = WMMA_BF16(a, ldgB(p2 + toff), c2[j]);
                    ca[j] = WMMA_BF16(a, ldgB(pa + toff), ca[j]);
                    cb[j] = WMMA_BF16(a, ldgB(pb + toff), cb[j]);
                }
            }
            float tsv[8];
            #pragma unroll
            for (int v = 0; v < 8; ++v) tsv[v] = tsrow[v + h8];
            #pragma unroll
            for (int j = 0; j < 2; ++j) {
                int nc = w * 32 + j * 16 + l15;   // h column 0..511
                float s1 = bias[1024 + nc];
                float s2 = bias[1536 + nc];
                float sa = bias[2048 + nc];
                float sb = bias[2560 + nc];
                #pragma unroll
                for (int v = 0; v < 8; ++v) {
                    int m = v + h8;
                    float f1  = tanh_fast(c1[j][v] + s1);
                    float f2v = tanh_fast(c2[j][v] + s2);
                    float ti  = sigmoid_fast((ca[j][v] + sa) * tsv[v] + (cb[j][v] + sb));
                    float hn  = f1 * (1.0f - ti) + ti * f2v;
                    Abuf[m * AB_STRIDE + 256 + nc] = f2bf(hn);
                }
            }
        }
        // stage barrier at top of next iteration orders h-writes vs. L0 reads
    }
    __syncthreads();

    // ---- output: h_last(16x512) @ W_out(512x128) + b_out -> f32 out ----
    if (w < 8) {
        v8f c = vzero;
        const __bf16* arow = Abuf + l15 * AB_STRIDE + 256 + h8;
        const __bf16* wp   = wout + opaque_i(((w * 16) << 9) + (lane << 4));
        #pragma unroll
        for (int kt = 0; kt < 16; ++kt) {
            v16bf a = ldsA(arow + kt * 32);
            v16bf b = ldgB(wp + (kt << 9));
            c = WMMA_BF16(a, b, c);
        }
        float bs = bias[3072 + w * 16 + l15];
        #pragma unroll
        for (int v = 0; v < 8; ++v)
            out[(size_t)(b0 + v + h8) * OUTD + w * 16 + l15] = c[v] + bs;
    }
}

// ---------------------------------------------------------------------------
extern "C" void kernel_launch(void* const* d_in, const int* in_sizes, int n_in,
                              void* d_out, int out_size, void* d_ws, size_t ws_size,
                              hipStream_t stream)
{
    (void)in_sizes; (void)n_in; (void)out_size; (void)ws_size;

    const float* x    = (const float*)d_in[0];
    const float* ts   = (const float*)d_in[1];
    const float* Wb0  = (const float*)d_in[2];
    const float* bb0  = (const float*)d_in[3];
    const float* Wbs  = (const float*)d_in[4];
    const float* bbs  = (const float*)d_in[5];
    const float* Wff1 = (const float*)d_in[6];
    const float* bff1 = (const float*)d_in[7];
    const float* Wff2 = (const float*)d_in[8];
    const float* bff2 = (const float*)d_in[9];
    const float* Wta  = (const float*)d_in[10];
    const float* bta  = (const float*)d_in[11];
    const float* Wtb  = (const float*)d_in[12];
    const float* btb  = (const float*)d_in[13];
    const float* Wout = (const float*)d_in[14];
    const float* bout = (const float*)d_in[15];

    __bf16* ws = (__bf16*)d_ws;
    const size_t OFF_WB0 = 0;            // 768*256
    const size_t OFF_WBS = 196608;       // 3 * 256*256
    const size_t OFF_F1  = 393216;       // 256*512
    const size_t OFF_F2  = 524288;
    const size_t OFF_TA  = 655360;
    const size_t OFF_TB  = 786432;
    const size_t OFF_OUT = 917504;       // 512*128  (total 983040 bf16 = 1.92 MB)

    auto pk = [&](const float* s, __bf16* d, int K, int N) {
        int tot = K * N;
        pack_bfrag<<<(tot + 255) / 256, 256, 0, stream>>>(s, d, N, K / 32, tot);
    };

    pk(Wb0, ws + OFF_WB0, 768, 256);
    for (int i = 0; i < 3; ++i)
        pk(Wbs + (size_t)i * 256 * 256, ws + OFF_WBS + (size_t)i * 65536, 256, 256);
    pk(Wff1, ws + OFF_F1, 256, 512);
    pk(Wff2, ws + OFF_F2, 256, 512);
    pk(Wta,  ws + OFF_TA, 256, 512);
    pk(Wtb,  ws + OFF_TB, 256, 512);
    pk(Wout, ws + OFF_OUT, 512, 128);

    cfc_scan_kernel<<<16, 512, 0, stream>>>(
        x, ts,
        ws + OFF_WB0, ws + OFF_WBS, ws + OFF_F1, ws + OFF_F2,
        ws + OFF_TA, ws + OFF_TB, ws + OFF_OUT,
        bb0, bbs, bff1, bff2, bta, btb, bout,
        (float*)d_out);
}
